// DeepseekV3Attention_52785148067898
// MI455X (gfx1250) — matrix operand order repaired and checked by reading, not compiled, not run on hardware
//
#include <hip/hip_runtime.h>

// ---------------------------------------------------------------------------
// DeepSeek-V3 MLA attention layer for gfx1250 (MI455X), bf16 WMMA pipeline.
// S=2048, HID=2048, NH=16, HEAD=192 (128 nope + 64 rope), VDIM=128.
// ---------------------------------------------------------------------------

typedef __bf16 bf16;
typedef bf16  v16bf __attribute__((ext_vector_type(16)));
typedef bf16  v8bf  __attribute__((ext_vector_type(8)));
typedef float v8f   __attribute__((ext_vector_type(8)));

#define S_SEQ   2048
#define HID     2048
#define NH      16
#define NOPE    128
#define ROPE    64
#define HEAD    192
#define VDIM    128
// SOFTMAX_SCALE = 192^-0.5 * (0.1*ln(40)+1)^2
#define SOFTMAX_SCALE 0.13523383f

static __device__ inline bf16 f2bf(float f) {
    unsigned int u = __builtin_bit_cast(unsigned int, f);
    u = u + 0x7FFFu + ((u >> 16) & 1u);
    unsigned short s = (unsigned short)(u >> 16);
    return __builtin_bit_cast(bf16, s);
}

static __device__ inline v8f wmma_bf16(v16bf a, v16bf b, v8f c) {
    // D = A(16x32 bf16) * B(32x16 bf16) + C(16x16 f32)
    return __builtin_amdgcn_wmma_f32_16x16x32_bf16(
        /*neg_a=*/false, a, /*neg_b=*/false, b,
        /*c_mod=*/(short)0, c, /*reuse_a=*/false, /*reuse_b=*/false);
}

// load 16 bf16 as two contiguous 16B chunks (b128 loads)
static __device__ inline v16bf load2x8(const bf16* p0, const bf16* p1) {
    v16bf r;
    ((v8bf*)&r)[0] = *(const v8bf*)p0;
    ((v8bf*)&r)[1] = *(const v8bf*)p1;
    return r;
}

// ---------------------------------------------------------------------------
// Elementwise prep kernels
// ---------------------------------------------------------------------------
__global__ void k_cvt_bf16(const float* __restrict__ x, bf16* __restrict__ y, long n) {
    long i = (long)blockIdx.x * 256 + threadIdx.x;
    if (i < n) y[i] = f2bf(x[i]);
}

// dequantize W (N,K) with block-128x128 scales s (ceil(N/128), sK) -> bf16
__global__ void k_dequant_bf16(const float* __restrict__ w, const float* __restrict__ s,
                               bf16* __restrict__ o, int N, int K, int sK) {
    long i = (long)blockIdx.x * 256 + threadIdx.x;
    if (i >= (long)N * K) return;
    int n = (int)(i / K), k = (int)(i % K);
    o[i] = f2bf(w[i] * s[(n >> 7) * sK + (k >> 7)]);
}

// row-wise RMSNorm: x f32 (rows, ld), take first `cols`, out bf16 (rows, cols)
__global__ void k_rmsnorm_bf16(const float* __restrict__ x, const float* __restrict__ w,
                               bf16* __restrict__ y, int cols, int ld) {
    int row = blockIdx.x;
    const float* xr = x + (size_t)row * ld;
    __shared__ float red[256];
    float ss = 0.f;
    for (int c = threadIdx.x; c < cols; c += 256) { float v = xr[c]; ss += v * v; }
    red[threadIdx.x] = ss;
    __syncthreads();
    for (int st = 128; st > 0; st >>= 1) {
        if ((int)threadIdx.x < st) red[threadIdx.x] += red[threadIdx.x + st];
        __syncthreads();
    }
    float inv = rsqrtf(red[0] / cols + 1e-6f);
    bf16* yr = y + (size_t)row * cols;
    for (int c = threadIdx.x; c < cols; c += 256) yr[c] = f2bf(xr[c] * inv * w[c]);
}

// q f32 (S, NH*HEAD) -> qcat bf16 same layout with RoPE on d in [128,192)
__global__ void k_q_prep(const float* __restrict__ q, const float* __restrict__ cosb,
                         const float* __restrict__ sinb, bf16* __restrict__ qcat) {
    long i = (long)blockIdx.x * 256 + threadIdx.x;
    if (i >= (long)S_SEQ * NH * HEAD) return;
    int s  = (int)(i / (NH * HEAD));
    int hd = (int)(i % (NH * HEAD));
    int d  = hd % HEAD;
    float v = q[i], o;
    if (d < NOPE) {
        o = v;
    } else {
        int dd = d - NOPE;
        float rot = (dd < 32) ? -q[i + 32] : q[i - 32];
        o = v * cosb[(size_t)s * ROPE + dd] + rot * sinb[(size_t)s * ROPE + dd];
    }
    qcat[i] = f2bf(o);
}

// kvb f32 (S, NH*256) -> knope bf16 (S,NH,128) and vT bf16 (NH*128, S)
__global__ void k_kv_prep(const float* __restrict__ kvb, bf16* __restrict__ knope,
                          bf16* __restrict__ vT) {
    long i = (long)blockIdx.x * 256 + threadIdx.x;
    if (i >= (long)S_SEQ * NH * VDIM) return;
    int t  = (int)(i >> 11);       // / 2048
    int hd = (int)(i & 2047);
    int h  = hd >> 7, d = hd & 127;
    knope[((size_t)t * NH + h) * NOPE + d] = f2bf(kvb[(size_t)t * (NH * 256) + h * 256 + d]);
    vT[(size_t)hd * S_SEQ + t]             = f2bf(kvb[(size_t)t * (NH * 256) + h * 256 + NOPE + d]);
}

// lkv f32 (S,576): RoPE the last 64 columns -> kpe bf16 (S,64)
__global__ void k_kpe_prep(const float* __restrict__ lkv, const float* __restrict__ cosb,
                           const float* __restrict__ sinb, bf16* __restrict__ kpe) {
    long i = (long)blockIdx.x * 256 + threadIdx.x;
    if (i >= (long)S_SEQ * ROPE) return;
    int s = (int)(i >> 6), d = (int)(i & 63);
    const float* row = lkv + (size_t)s * 576 + 512;
    float x   = row[d];
    float rot = (d < 32) ? -row[d + 32] : row[d - 32];
    kpe[i] = f2bf(x * cosb[i] + rot * sinb[i]);
}

// ---------------------------------------------------------------------------
// Register-blocked, double-buffered (2x manually unrolled) bf16 WMMA GEMM:
//   C(M,N) f32 = A(M,K) bf16 @ W(N,K)^T bf16
// Each wave computes a 32x64 tile (2 M-subtiles x 4 N-subtiles -> 8 WMMAs per
// K=32 step, 1.5 b128 loads/WMMA). Two named buffer sets ping-pong so there
// are no register-rotation copies: while set0 is computed, set1's loads are
// in flight, and vice versa. Requires K % 64 == 0 (true for K=2048/1536/512).
// Block = 256 threads = 8 waves (2x4) -> 64x256/block.
// grid = (ceil(N/256), M/64); per-wave nbase guard handles N=576.
// ---------------------------------------------------------------------------
__global__ __launch_bounds__(256)
void k_gemm_bf16_wmma(const bf16* __restrict__ A, const bf16* __restrict__ W,
                      float* __restrict__ C, int M, int N, int K) {
    const int lane = threadIdx.x & 31;
    const int wid  = threadIdx.x >> 5;
    const int half = lane >> 4;
    const int l16  = lane & 15;
    const int mbase = blockIdx.y * 64 + (wid >> 2) * 32;
    const int nbase = blockIdx.x * 256 + (wid & 3) * 64;
    if (nbase >= N) return;   // wave-uniform guard for ragged N

    const bf16* arow0 = A + (size_t)(mbase + l16) * K;
    const bf16* arow1 = A + (size_t)(mbase + 16 + l16) * K;
    const bf16* wrow[4];
#pragma unroll
    for (int j = 0; j < 4; ++j) wrow[j] = W + (size_t)(nbase + j * 16 + l16) * K;

    const int khA = half * 8;   // A: lanes 0-15 hold K 0-7/16-23, lanes 16-31 hold 8-15/24-31
    const int khB = half * 8;   // B: lanes 0-15 hold K 0-7/16-23, lanes 16-31 hold 8-15/24-31

    v8f acc[2][4];
#pragma unroll
    for (int i = 0; i < 2; ++i)
#pragma unroll
        for (int j = 0; j < 4; ++j) acc[i][j] = (v8f){};

    v16bf aS0[2], aS1[2];
    v16bf bS0[4], bS1[4];

    auto load_set = [&](v16bf* aB, v16bf* bB, int k) {
        aB[0] = load2x8(arow0 + k + khA, arow0 + k + 16 + khA);
        aB[1] = load2x8(arow1 + k + khA, arow1 + k + 16 + khA);
#pragma unroll
        for (int j = 0; j < 4; ++j)
            bB[j] = load2x8(wrow[j] + k + khB, wrow[j] + k + 16 + khB);
    };
    auto compute_set = [&](const v16bf* aB, const v16bf* bB) {
#pragma unroll
        for (int j = 0; j < 4; ++j) {
            acc[0][j] = wmma_bf16(aB[0], bB[j], acc[0][j]);
            acc[1][j] = wmma_bf16(aB[1], bB[j], acc[1][j]);
        }
    };

    // ---- prologue: fill both buffer sets ----
    load_set(aS0, bS0, 0);
    load_set(aS1, bS1, 32);

    // ---- steady state (K % 64 == 0): compute Si, refill Si for k0+... ----
    for (int k0 = 64; k0 < K; k0 += 64) {
        __builtin_prefetch(arow0 + k0 + 512, 0, 1);   // global_prefetch_b8
        __builtin_prefetch(wrow[0] + k0 + 512, 0, 1);
        compute_set(aS0, bS0);
        load_set(aS0, bS0, k0);
        compute_set(aS1, bS1);
        load_set(aS1, bS1, k0 + 32);
    }

    // ---- epilogue: last two K-steps ----
    compute_set(aS0, bS0);
    compute_set(aS1, bS1);

#pragma unroll
    for (int i = 0; i < 2; ++i) {
        float* crow = C + (size_t)(mbase + i * 16 + 8 * half) * N + nbase + l16;
#pragma unroll
        for (int j = 0; j < 4; ++j)
#pragma unroll
            for (int r = 0; r < 8; ++r)
                crow[(size_t)r * N + j * 16] = acc[i][j][r];
    }
}

// ---------------------------------------------------------------------------
// Flash attention: one wave per (head, 16-query tile). Online softmax, keys in
// tiles of 32 (two 16x16 score WMMAs -> one 16x32 P used as A for P@V).
// P is re-laid-out C-layout -> A-layout through per-wave LDS.
// ---------------------------------------------------------------------------
__global__ __launch_bounds__(128)
void k_mla_attention(const bf16* __restrict__ qcat, const bf16* __restrict__ knope,
                     const bf16* __restrict__ kpe, const bf16* __restrict__ vT,
                     bf16* __restrict__ out) {
    const int lane = threadIdx.x & 31;
    const int wid  = threadIdx.x >> 5;
    const int half = lane >> 4;
    const int l16  = lane & 15;
    const int gw   = blockIdx.x * 4 + wid;   // 2048 waves total
    const int qt   = gw & 127;               // query tile (S/16 = 128)
    const int h    = gw >> 7;                // head
    const int qbase = qt * 16;

    __shared__ bf16 pst[4][16][32];          // per-wave P staging (C->A relayout)
    bf16 (*pw)[32] = pst[wid];

    const int khA = half * 8;
    const int khB = half * 8;

    // Q tile (16 x 192) as six K=32 A-operands, loaded once
    const bf16* qrow = qcat + ((size_t)(qbase + l16) * NH + h) * HEAD;
    v16bf qa[6];
#pragma unroll
    for (int c = 0; c < 6; ++c)
        qa[c] = load2x8(qrow + c * 32 + khA, qrow + c * 32 + 16 + khA);

    v8f acc[8];
#pragma unroll
    for (int ct = 0; ct < 8; ++ct) acc[ct] = (v8f){};
    float mrow[8], lrow[8];
#pragma unroll
    for (int r = 0; r < 8; ++r) { mrow[r] = -1e30f; lrow[r] = 0.f; }

    const int nt = (qbase + 47) >> 5;        // 32-key tiles covering [0, qbase+15]
    for (int kt = 0; kt < nt; ++kt) {
        const int tbase = kt * 32;

        // ---- scores: two 16-key column tiles ----
        v8f sc[2];
#pragma unroll
        for (int tt = 0; tt < 2; ++tt) {
            const int tcol = tbase + tt * 16 + l16;   // B-operand key index for this lane
            v8f s = {};
#pragma unroll
            for (int c = 0; c < 6; ++c) {
                const bf16* kp = (c < 4)
                    ? knope + ((size_t)tcol * NH + h) * NOPE + c * 32 + khB
                    : kpe + (size_t)tcol * ROPE + (c - 4) * 32 + khB;
                v16bf b = load2x8(kp, kp + 16);
                s = wmma_bf16(qa[c], b, s);
            }
            sc[tt] = s;
        }

        // ---- online softmax (rows live across lane-halves; width-16 shuffles) ----
#pragma unroll
        for (int r = 0; r < 8; ++r) {
            const int srow = qbase + r + 8 * half;
            float s0 = sc[0][r] * SOFTMAX_SCALE;
            float s1 = sc[1][r] * SOFTMAX_SCALE;
            if (tbase + l16 > srow)      s0 = -1e30f;
            if (tbase + 16 + l16 > srow) s1 = -1e30f;
            float mx = fmaxf(s0, s1);
#pragma unroll
            for (int d = 1; d < 16; d <<= 1) mx = fmaxf(mx, __shfl_xor(mx, d, 16));
            float mnew  = fmaxf(mrow[r], mx);
            float alpha = __expf(mrow[r] - mnew);
            float p0 = __expf(s0 - mnew);
            float p1 = __expf(s1 - mnew);
            float rs = p0 + p1;
#pragma unroll
            for (int d = 1; d < 16; d <<= 1) rs += __shfl_xor(rs, d, 16);
            lrow[r] = lrow[r] * alpha + rs;
            mrow[r] = mnew;
#pragma unroll
            for (int ct = 0; ct < 8; ++ct) acc[ct][r] *= alpha;
            // stage P (C layout -> LDS)
            pw[r + 8 * half][l16]      = f2bf(p0);
            pw[r + 8 * half][16 + l16] = f2bf(p1);
        }
        asm volatile("s_wait_dscnt 0" ::: "memory");

        // reload P in A layout (16x32)
        const bf16* prow = &pw[l16][0];
        v16bf pa = load2x8(prow + khA, prow + 16 + khA);

        // ---- P @ V : 8 column tiles of VDIM, vT is (NH*128, S) so K is contiguous ----
#pragma unroll
        for (int ct = 0; ct < 8; ++ct) {
            const bf16* vp = vT + ((size_t)h * VDIM + ct * 16 + l16) * S_SEQ + tbase + khB;
            v16bf vb = load2x8(vp, vp + 16);
            acc[ct] = wmma_bf16(pa, vb, acc[ct]);
        }
    }

    // epilogue: normalize and write attn (S, NH*VDIM) bf16
#pragma unroll
    for (int ct = 0; ct < 8; ++ct) {
#pragma unroll
        for (int r = 0; r < 8; ++r) {
            float o = acc[ct][r] / lrow[r];
            out[(size_t)(qbase + r + 8 * half) * (NH * VDIM) + h * VDIM + ct * 16 + l16] = f2bf(o);
        }
    }
}

// ---------------------------------------------------------------------------
// Host-side launch
// ---------------------------------------------------------------------------
extern "C" void kernel_launch(void* const* d_in, const int* in_sizes, int n_in,
                              void* d_out, int out_size, void* d_ws, size_t ws_size,
                              hipStream_t stream) {
    (void)in_sizes; (void)n_in; (void)out_size; (void)ws_size;
    const float* hs    = (const float*)d_in[0];
    const float* wq_a  = (const float*)d_in[1];
    const float* sq_a  = (const float*)d_in[2];
    const float* wq_b  = (const float*)d_in[3];
    const float* sq_b  = (const float*)d_in[4];
    const float* wkv_a = (const float*)d_in[5];
    const float* skv_a = (const float*)d_in[6];
    const float* wkv_b = (const float*)d_in[7];
    const float* skv_b = (const float*)d_in[8];
    const float* wo    = (const float*)d_in[9];
    const float* so    = (const float*)d_in[10];
    const float* q_ln  = (const float*)d_in[11];
    const float* kv_ln = (const float*)d_in[12];
    const float* cosb  = (const float*)d_in[13];
    const float* sinb  = (const float*)d_in[14];
    float* outp = (float*)d_out;

    char* wsb = (char*)d_ws;
    size_t off = 0;
    auto alloc = [&](size_t bytes) -> char* {
        char* p = wsb + off;
        off += (bytes + 255) & ~(size_t)255;
        return p;
    };

    bf16*  xb     = (bf16*)alloc((size_t)S_SEQ * HID * 2);          // hs bf16
    bf16*  wqa_b  = (bf16*)alloc((size_t)1536 * 2048 * 2);
    bf16*  wqb_b  = (bf16*)alloc((size_t)3072 * 1536 * 2);
    bf16*  wkva_b = (bf16*)alloc((size_t)576 * 2048 * 2);
    bf16*  wkvb_b = (bf16*)alloc((size_t)4096 * 512 * 2);
    bf16*  wo_b   = (bf16*)alloc((size_t)2048 * 2048 * 2);
    float* lq     = (float*)alloc((size_t)S_SEQ * 1536 * 4);
    bf16*  qn_b   = (bf16*)alloc((size_t)S_SEQ * 1536 * 2);
    float* qf     = (float*)alloc((size_t)S_SEQ * 3072 * 4);
    float* lkv    = (float*)alloc((size_t)S_SEQ * 576 * 4);
    bf16*  kvn_b  = (bf16*)alloc((size_t)S_SEQ * 512 * 2);
    float* kvbf   = (float*)alloc((size_t)S_SEQ * 4096 * 4);
    bf16*  qcat   = (bf16*)alloc((size_t)S_SEQ * NH * HEAD * 2);
    bf16*  knope  = (bf16*)alloc((size_t)S_SEQ * NH * NOPE * 2);
    bf16*  kpe    = (bf16*)alloc((size_t)S_SEQ * ROPE * 2);
    bf16*  vTb    = (bf16*)alloc((size_t)NH * VDIM * S_SEQ * 2);
    bf16*  attnb  = (bf16*)alloc((size_t)S_SEQ * NH * VDIM * 2);

    auto g1 = [](long n) { return (unsigned)((n + 255) / 256); };
    auto ggemm = [](int M, int N) { return dim3((unsigned)((N + 255) / 256), (unsigned)(M / 64)); };

    // 0) activations + weights -> bf16 (scales folded)
    k_cvt_bf16<<<g1((long)S_SEQ * HID), 256, 0, stream>>>(hs, xb, (long)S_SEQ * HID);
    k_dequant_bf16<<<g1(1536L * 2048), 256, 0, stream>>>(wq_a,  sq_a,  wqa_b,  1536, 2048, 16);
    k_dequant_bf16<<<g1(3072L * 1536), 256, 0, stream>>>(wq_b,  sq_b,  wqb_b,  3072, 1536, 12);
    k_dequant_bf16<<<g1(576L  * 2048), 256, 0, stream>>>(wkv_a, skv_a, wkva_b, 576,  2048, 16);
    k_dequant_bf16<<<g1(4096L * 512),  256, 0, stream>>>(wkv_b, skv_b, wkvb_b, 4096, 512,  4);
    k_dequant_bf16<<<g1(2048L * 2048), 256, 0, stream>>>(wo,    so,    wo_b,   2048, 2048, 16);

    // 1) lq = hs @ wq_a^T ; rmsnorm -> qn_b
    k_gemm_bf16_wmma<<<ggemm(S_SEQ, 1536), 256, 0, stream>>>(xb, wqa_b, lq, S_SEQ, 1536, 2048);
    k_rmsnorm_bf16<<<S_SEQ, 256, 0, stream>>>(lq, q_ln, qn_b, 1536, 1536);

    // 2) q = qn @ wq_b^T
    k_gemm_bf16_wmma<<<ggemm(S_SEQ, 3072), 256, 0, stream>>>(qn_b, wqb_b, qf, S_SEQ, 3072, 1536);

    // 3) lkv = hs @ wkv_a^T ; rmsnorm(first 512) -> kvn_b
    k_gemm_bf16_wmma<<<ggemm(S_SEQ, 576), 256, 0, stream>>>(xb, wkva_b, lkv, S_SEQ, 576, 2048);
    k_rmsnorm_bf16<<<S_SEQ, 256, 0, stream>>>(lkv, kv_ln, kvn_b, 512, 576);

    // 4) kvb = kvn @ wkv_b^T
    k_gemm_bf16_wmma<<<ggemm(S_SEQ, 4096), 256, 0, stream>>>(kvn_b, wkvb_b, kvbf, S_SEQ, 4096, 512);

    // 5) RoPE / repack
    k_q_prep<<<g1((long)S_SEQ * NH * HEAD), 256, 0, stream>>>(qf, cosb, sinb, qcat);
    k_kv_prep<<<g1((long)S_SEQ * NH * VDIM), 256, 0, stream>>>(kvbf, knope, vTb);
    k_kpe_prep<<<g1((long)S_SEQ * ROPE), 256, 0, stream>>>(lkv, cosb, sinb, kpe);

    // 6) flash attention (2048 waves; 4 waves per block)
    k_mla_attention<<<(S_SEQ / 16) * NH / 4, 128, 0, stream>>>(qcat, knope, kpe, vTb, attnb);

    // 7) out = attn @ wo^T  (f32 straight into d_out)
    k_gemm_bf16_wmma<<<ggemm(S_SEQ, 2048), 256, 0, stream>>>(attnb, wo_b, outp, S_SEQ, 2048, 2048);
}
